// LRA_GNN_89550068122089
// MI455X (gfx1250) — compile-verified, matrix operand order checked by reading, not run on hardware
//
#include <hip/hip_runtime.h>
#include <hip/hip_bf16.h>

// ---------------------------------------------------------------------------
// Problem constants (match the JAX reference)
// ---------------------------------------------------------------------------
#define NNODES 4096
#define NEDGES 131072
#define INC    512
#define HID    512
#define HEADS  4
#define LAYERS 3
#define NGRAPH 16
#define NEGSL  0.2f
#define HD     (HEADS * HID)   // 2048

// ---------------------------------------------------------------------------
// CDNA5 feature detection (compile-safe: fall back if builtins absent)
// ---------------------------------------------------------------------------
#ifndef __has_builtin
#define __has_builtin(x) 0
#endif
#if __has_builtin(__builtin_amdgcn_global_load_async_to_lds_b128) && \
    __has_builtin(__builtin_amdgcn_s_wait_asynccnt)
#define USE_ASYNC_LDS 1
#else
#define USE_ASYNC_LDS 0
#endif

// ---------------------------------------------------------------------------
// WMMA / async types (CDNA5 / gfx1250, wave32)
// ---------------------------------------------------------------------------
typedef __attribute__((ext_vector_type(16))) __bf16 v16bf;
typedef __attribute__((ext_vector_type(8)))  float  v8f;
typedef __attribute__((__vector_size__(16))) int    v4i_n;   // native 128-bit vec
typedef __attribute__((address_space(1))) v4i_n*    gptr128_t;
typedef __attribute__((address_space(3))) v4i_n*    lptr128_t;

union FragAB { v16bf v; uint4 q[2]; };

// ---------------------------------------------------------------------------
// Tiled bf16 x bf16 -> f32 GEMM via V_WMMA_F32_16X16X32_BF16.
//   C[M,N] = A[M,K] @ B[K,N],  with B supplied PRE-TRANSPOSED as BT[N,K]
//   (so both A and BT tiles stream row-major: b128-only global & LDS traffic).
// Block tile 128x128, K step 32, 256 threads = 8 wave32 waves in a 4(M)x2(N)
// grid; each wave owns a 32x64 tile = 2x4 WMMA fragments (8 WMMAs / K-step).
// Double-buffered LDS, software-pipelined; async global->LDS (ASYNCcnt) when
// the toolchain exposes it, else register-staged prefetch.
// Requires M%128==0, N%128==0, K%32==0 (true for all launches here).
// ---------------------------------------------------------------------------
#define GEMM_TM 128
#define GEMM_TN 128
#define GEMM_TK 32
#define LDSW    40   // 32 + 8 pad halves; 80B row stride stays 16B-aligned

__global__ __launch_bounds__(256)
void gemm_bf16_wmma(const unsigned short* __restrict__ Abf,
                    const unsigned short* __restrict__ BT,   // [N][K] row-major
                    float* __restrict__ C,
                    const float* __restrict__ Resid,
                    int M, int N, int K,
                    int lda, int ldbT, int ldc,
                    long b_zstride, int c_zcoloff,
                    int do_relu) {
  __shared__ unsigned short sA[2][GEMM_TM * LDSW];   // 2 x 10240 B
  __shared__ unsigned short sB[2][GEMM_TN * LDSW];   // 2 x 10240 B

  const int tid      = threadIdx.x;
  const int lane     = tid & 31;
  const int wave     = tid >> 5;
  const int wm       = wave >> 1;     // 0..3  (M)
  const int wn       = wave & 1;      // 0..1  (N)
  const int laneHalf = lane >> 4;     // 0/1
  const int laneR    = lane & 15;

  const int  blockM   = blockIdx.y * GEMM_TM;
  const int  blockN   = blockIdx.x * GEMM_TN;
  const int  z        = blockIdx.z;
  const unsigned short* Bz = BT + (long)z * b_zstride;
  const long cColBase = (long)z * c_zcoloff + blockN;

  // Each 128x32-half tile = 512 chunks of 8 halves; 256 threads take 2 each.
  const int row0 = tid >> 2;               const int cc0 = (tid & 3) << 3;
  const int row1 = (tid + 256) >> 2;       const int cc1 = ((tid + 256) & 3) << 3;
  const unsigned short* gA0 = Abf + (long)(blockM + row0) * lda  + cc0;
  const unsigned short* gA1 = Abf + (long)(blockM + row1) * lda  + cc1;
  const unsigned short* gB0 = Bz  + (long)(blockN + row0) * ldbT + cc0;
  const unsigned short* gB1 = Bz  + (long)(blockN + row1) * ldbT + cc1;
  const int lA0 = row0 * LDSW + cc0;
  const int lA1 = row1 * LDSW + cc1;

  v8f acc[2][4] = {};

#if USE_ASYNC_LDS
  auto asyncCopy = [&](int buf, int k0) {
    __builtin_amdgcn_global_load_async_to_lds_b128(
        (gptr128_t)(void*)(gA0 + k0), (lptr128_t)(void*)&sA[buf][lA0], 0, 0);
    __builtin_amdgcn_global_load_async_to_lds_b128(
        (gptr128_t)(void*)(gA1 + k0), (lptr128_t)(void*)&sA[buf][lA1], 0, 0);
    __builtin_amdgcn_global_load_async_to_lds_b128(
        (gptr128_t)(void*)(gB0 + k0), (lptr128_t)(void*)&sB[buf][lA0], 0, 0);
    __builtin_amdgcn_global_load_async_to_lds_b128(
        (gptr128_t)(void*)(gB1 + k0), (lptr128_t)(void*)&sB[buf][lA1], 0, 0);
  };
#else
  uint4 rA0, rA1, rB0, rB1;
  auto loadRegs = [&](int k0) {
    rA0 = *(const uint4*)(gA0 + k0);
    rA1 = *(const uint4*)(gA1 + k0);
    rB0 = *(const uint4*)(gB0 + k0);
    rB1 = *(const uint4*)(gB1 + k0);
  };
  auto storeRegs = [&](int buf) {
    *(uint4*)(&sA[buf][lA0]) = rA0;
    *(uint4*)(&sA[buf][lA1]) = rA1;
    *(uint4*)(&sB[buf][lA0]) = rB0;
    *(uint4*)(&sB[buf][lA1]) = rB1;
  };
#endif

  auto computeTile = [&](int cur) {
    // Fragment loads per documented 16-bit 16x32 layout:
    // lane r (half 0) holds K [0..7]+[16..23]; lane r+16 holds K [8..15]+[24..31]
    FragAB a[2], b[4];
#pragma unroll
    for (int m = 0; m < 2; ++m) {
      const unsigned short* base = &sA[cur][(wm * 32 + m * 16 + laneR) * LDSW];
      a[m].q[0] = *(const uint4*)(base + laneHalf * 8);
      a[m].q[1] = *(const uint4*)(base + 16 + laneHalf * 8);
    }
#pragma unroll
    for (int n = 0; n < 4; ++n) {
      const unsigned short* base = &sB[cur][(wn * 64 + n * 16 + laneR) * LDSW];
      b[n].q[0] = *(const uint4*)(base + laneHalf * 8);
      b[n].q[1] = *(const uint4*)(base + 16 + laneHalf * 8);
    }
#pragma unroll
    for (int m = 0; m < 2; ++m)
#pragma unroll
      for (int n = 0; n < 4; ++n)
        acc[m][n] = __builtin_amdgcn_wmma_f32_16x16x32_bf16(
            false, a[m].v, false, b[n].v, (short)0, acc[m][n], false, false);
  };

  const int nT = K / GEMM_TK;

#if USE_ASYNC_LDS
  asyncCopy(0, 0);
  __builtin_amdgcn_s_wait_asynccnt(0);
  __syncthreads();
  for (int t = 0; t < nT; ++t) {
    const int cur = t & 1;
    if (t + 1 < nT) asyncCopy(1 - cur, (t + 1) * GEMM_TK);
    computeTile(cur);
    __builtin_amdgcn_s_wait_asynccnt(0);
    __syncthreads();
  }
#else
  loadRegs(0);
  storeRegs(0);
  __syncthreads();
  for (int t = 0; t < nT; ++t) {
    const int cur = t & 1;
    if (t + 1 < nT) loadRegs((t + 1) * GEMM_TK);   // issue global loads
    computeTile(cur);                              // overlap with loads
    if (t + 1 < nT) storeRegs(1 - cur);
    __syncthreads();
  }
#endif

  // Store per documented 16x16 f32 C layout:
  // lanes 0-15 VGPR v -> M=v ; lanes 16-31 VGPR v -> M=8+v ; N = lane&15
#pragma unroll
  for (int m = 0; m < 2; ++m) {
    int rowBase = blockM + wm * 32 + m * 16 + laneHalf * 8;
#pragma unroll
    for (int n = 0; n < 4; ++n) {
      long col = cColBase + wn * 64 + n * 16 + laneR;
#pragma unroll
      for (int v = 0; v < 8; ++v) {
        long idx  = (long)(rowBase + v) * ldc + col;
        float val = acc[m][n][v];
        if (do_relu) val = fmaxf(val, 0.0f);
        if (Resid)   val += Resid[idx];
        C[idx] = val;
      }
    }
  }
}

// ---------------------------------------------------------------------------
// Helpers / small kernels
// ---------------------------------------------------------------------------
__device__ __forceinline__ unsigned short f32_to_bf16_rne(float f) {
  unsigned int u = __float_as_uint(f);
  unsigned int r = u + 0x7FFFu + ((u >> 16) & 1u);
  return (unsigned short)(r >> 16);
}

__global__ void cvt_f32_bf16_kernel(const float* __restrict__ in,
                                    unsigned short* __restrict__ out, long n) {
  long i = (long)blockIdx.x * blockDim.x + threadIdx.x;
  if (i < n) out[i] = f32_to_bf16_rne(in[i]);
}

// batched transpose-convert: in[z][R][C] (f32) -> out[z][C][R] (bf16)
__global__ void cvt_f32_bf16_T_kernel(const float* __restrict__ in,
                                      unsigned short* __restrict__ out,
                                      int R, int C, long total) {
  long i = (long)blockIdx.x * blockDim.x + threadIdx.x;
  if (i >= total) return;
  long rc   = (long)R * C;
  long z    = i / rc;
  long rem  = i - z * rc;
  int  r    = (int)(rem / C);
  int  c    = (int)(rem - (long)r * C);
  out[z * rc + (long)c * R + r] = f32_to_bf16_rne(in[i]);
}

__global__ void fill_f32_kernel(float* __restrict__ p, float v, long n) {
  long i = (long)blockIdx.x * blockDim.x + threadIdx.x;
  if (i < n) p[i] = v;
}

__device__ __forceinline__ void atomicMaxFloat(float* addr, float val) {
  unsigned int* ua = (unsigned int*)addr;
  unsigned int old = *ua;
  while (__uint_as_float(old) < val) {
    unsigned int assumed = old;
    old = atomicCAS(ua, assumed, __float_as_uint(val));
    if (old == assumed) break;
  }
}

__device__ __forceinline__ float blockReduceSum256(float v, float* sbuf) {
  int tid = threadIdx.x;
  sbuf[tid] = v; __syncthreads();
  for (int s = blockDim.x >> 1; s > 0; s >>= 1) {
    if (tid < s) sbuf[tid] += sbuf[tid + s];
    __syncthreads();
  }
  return sbuf[0];
}

// lg_src[n,h] = <h[n,h,:], a_src[h,:]> ; same for dst. One block per (n,h).
__global__ void lg_kernel(const float* __restrict__ h,
                          const float* __restrict__ a_src,
                          const float* __restrict__ a_dst,
                          float* __restrict__ lgs, float* __restrict__ lgd) {
  __shared__ float sbuf[128];
  int nh = blockIdx.x;               // n*HEADS + head
  int head = nh & (HEADS - 1);
  const float* hp = h + (long)nh * HID;
  float s = 0.f, d = 0.f;
  for (int i = threadIdx.x; i < HID; i += blockDim.x) {
    float v = hp[i];
    s += v * a_src[head * HID + i];
    d += v * a_dst[head * HID + i];
  }
  float S = blockReduceSum256(s, sbuf);
  __syncthreads();
  float D = blockReduceSum256(d, sbuf);
  if (threadIdx.x == 0) { lgs[nh] = S; lgd[nh] = D; }
}

// e[e,h] = leaky_relu(lg_src[src]+lg_dst[dst]); running segment max into m[dst,h]
__global__ void edge_e_kernel(const int* __restrict__ src, const int* __restrict__ dst,
                              const float* __restrict__ lgs, const float* __restrict__ lgd,
                              float* __restrict__ earr, float* __restrict__ m) {
  int i = blockIdx.x * blockDim.x + threadIdx.x;
  if (i >= NEDGES) return;
  int s = src[i], d = dst[i];
#pragma unroll
  for (int h = 0; h < HEADS; ++h) {
    float v = lgs[s * HEADS + h] + lgd[d * HEADS + h];
    v = (v > 0.f) ? v : NEGSL * v;
    earr[i * HEADS + h] = v;
    atomicMaxFloat(&m[d * HEADS + h], v);
  }
}

// w = exp(e - m[dst]); z[dst,h] += w (in-place overwrite of earr with w)
__global__ void edge_w_kernel(const int* __restrict__ dst,
                              float* __restrict__ earr,
                              const float* __restrict__ m,
                              float* __restrict__ z) {
  int i = blockIdx.x * blockDim.x + threadIdx.x;
  if (i >= NEDGES) return;
  int d = dst[i];
#pragma unroll
  for (int h = 0; h < HEADS; ++h) {
    float w = __expf(earr[i * HEADS + h] - m[d * HEADS + h]);
    earr[i * HEADS + h] = w;
    atomicAdd(&z[d * HEADS + h], w);
  }
}

// H0[dst,h,d] += (w/(z+eps)) * hfeat[src,h,d]  (one block per edge)
__global__ void edge_scatter_kernel(const int* __restrict__ src, const int* __restrict__ dst,
                                    const float* __restrict__ w, const float* __restrict__ z,
                                    const float* __restrict__ hfeat, float* __restrict__ H0) {
  int e = blockIdx.x;
  int s = src[e], d = dst[e];
  for (int t = threadIdx.x; t < HD; t += blockDim.x) {
    int head = t >> 9;   // /HID
    float alpha = w[e * HEADS + head] / (z[d * HEADS + head] + 1e-16f);
    atomicAdd(&H0[(long)d * HD + t], alpha * hfeat[(long)s * HD + t]);
  }
}

// dense adjacency scatter: A[src,dst] = 1 (duplicates idempotent, like .set)
__global__ void scatter_A_kernel(const int* __restrict__ src, const int* __restrict__ dst,
                                 float* __restrict__ A) {
  int i = blockIdx.x * blockDim.x + threadIdx.x;
  if (i < NEDGES) A[(long)src[i] * NNODES + dst[i]] = 1.0f;
}

// add self loop, compute dinv[n] = rsqrt(rowsum). One block per row.
__global__ void row_deg_kernel(float* __restrict__ A, float* __restrict__ dinv) {
  __shared__ float sbuf[256];
  int n = blockIdx.x;
  if (threadIdx.x == 0) A[(long)n * NNODES + n] += 1.0f;
  __syncthreads();
  float s = 0.f;
  for (int j = threadIdx.x; j < NNODES; j += blockDim.x)
    s += A[(long)n * NNODES + j];
  float S = blockReduceSum256(s, sbuf);
  if (threadIdx.x == 0) dinv[n] = rsqrtf(S);
}

// A_hat(bf16)[i,j] = dinv[i]*A[i,j]*dinv[j]
__global__ void ahat_kernel(const float* __restrict__ A, const float* __restrict__ dinv,
                            unsigned short* __restrict__ Ahat) {
  long i = (long)blockIdx.x * blockDim.x + threadIdx.x;
  if (i >= (long)NNODES * NNODES) return;
  int r = (int)(i >> 12);           // /4096
  int c = (int)(i & (NNODES - 1));
  Ahat[i] = f32_to_bf16_rne(dinv[r] * A[i] * dinv[c]);
}

__global__ void cnt_kernel(const int* __restrict__ batch, float* __restrict__ cnt) {
  int i = blockIdx.x * blockDim.x + threadIdx.x;
  if (i < NNODES) atomicAdd(&cnt[batch[i]], 1.0f);
}

// pooled[g, h, d] += Hcur[n, h, d]  (one block per node)
__global__ void pool_kernel(const int* __restrict__ batch, const float* __restrict__ Hc,
                            float* __restrict__ pooled) {
  int n = blockIdx.x;
  int g = batch[n];
  for (int t = threadIdx.x; t < HD; t += blockDim.x)
    atomicAdd(&pooled[(long)g * HD + t], Hc[(long)n * HD + t]);
}

// out[g] = (1/(HEADS*cnt[g])) * sum_{h,d} pooled[g,h,d]*fc_w[d] + fc_b[0]
__global__ void final_kernel(const float* __restrict__ pooled, const float* __restrict__ cnt,
                             const float* __restrict__ fc_w, const float* __restrict__ fc_b,
                             float* __restrict__ out) {
  __shared__ float sbuf[256];
  int g = blockIdx.x;
  float acc = 0.f;
  for (int t = threadIdx.x; t < HD; t += blockDim.x)
    acc += pooled[(long)g * HD + t] * fc_w[t & (HID - 1)];
  float S = blockReduceSum256(acc, sbuf);
  if (threadIdx.x == 0) {
    float c = fmaxf(cnt[g], 1.0f);
    out[g] = S / ((float)HEADS * c) + fc_b[0];
  }
}

// ---------------------------------------------------------------------------
// Workspace layout (bytes). Total ~245 MB.
// ---------------------------------------------------------------------------
#define OFF_A       0L                                   // 64 MB f32 dense A
#define OFF_AHAT    (OFF_A      + 67108864L)             // 32 MB bf16 A_hat
#define OFF_H       (OFF_AHAT   + 33554432L)             // 32 MB f32 h [N,H,D]
#define OFF_HCUR    (OFF_H      + 33554432L)             // 32 MB f32 Hcur
#define OFF_HBFT    (OFF_HCUR   + 33554432L)             // 16 MB bf16 Hcur^T
#define OFF_P       (OFF_HBFT   + 16777216L)             // 32 MB f32 P
#define OFF_PBF     (OFF_P      + 33554432L)             // 16 MB bf16 P
#define OFF_XBF     (OFF_PBF    + 16777216L)             //  4 MB bf16 x
#define OFF_WGATT   (OFF_XBF    + 4194304L)              //  2 MB bf16 W_gat^T
#define OFF_WGCNT   (OFF_WGATT  + 2097152L)              // 1.5MB bf16 W_gcn^T
#define OFF_LGS     (OFF_WGCNT  + 1572864L)
#define OFF_LGD     (OFF_LGS    + 65536L)
#define OFF_M       (OFF_LGD    + 65536L)
#define OFF_Z       (OFF_M      + 65536L)
#define OFF_EW      (OFF_Z      + 65536L)                //  2 MB edge e/w
#define OFF_DINV    (OFF_EW     + 2097152L)
#define OFF_POOLED  (OFF_DINV   + 16384L)
#define OFF_CNT     (OFF_POOLED + 131072L)

extern "C" void kernel_launch(void* const* d_in, const int* in_sizes, int n_in,
                              void* d_out, int out_size, void* d_ws, size_t ws_size,
                              hipStream_t stream) {
  const float* x      = (const float*)d_in[0];
  const int*   ei     = (const int*)  d_in[1];
  const int*   batch  = (const int*)  d_in[2];
  const float* W_gat  = (const float*)d_in[3];
  const float* a_src  = (const float*)d_in[4];
  const float* a_dst  = (const float*)d_in[5];
  const float* W_gcn  = (const float*)d_in[6];
  const float* fc_w   = (const float*)d_in[7];
  const float* fc_b   = (const float*)d_in[8];
  const int* src = ei;
  const int* dst = ei + NEDGES;

  char* ws = (char*)d_ws;
  float*          A_f32   = (float*)(ws + OFF_A);
  unsigned short* Ahat    = (unsigned short*)(ws + OFF_AHAT);
  float*          h_f32   = (float*)(ws + OFF_H);
  float*          Hcur    = (float*)(ws + OFF_HCUR);
  unsigned short* HbfT    = (unsigned short*)(ws + OFF_HBFT);
  float*          P_f32   = (float*)(ws + OFF_P);
  unsigned short* Pbf     = (unsigned short*)(ws + OFF_PBF);
  unsigned short* xbf     = (unsigned short*)(ws + OFF_XBF);
  unsigned short* WgatT   = (unsigned short*)(ws + OFF_WGATT);
  unsigned short* WgcnT   = (unsigned short*)(ws + OFF_WGCNT);
  float*          lgs     = (float*)(ws + OFF_LGS);
  float*          lgd     = (float*)(ws + OFF_LGD);
  float*          mbuf    = (float*)(ws + OFF_M);
  float*          zbuf    = (float*)(ws + OFF_Z);
  float*          ew      = (float*)(ws + OFF_EW);
  float*          dinv    = (float*)(ws + OFF_DINV);
  float*          pooled  = (float*)(ws + OFF_POOLED);
  float*          cnt     = (float*)(ws + OFF_CNT);
  float*          out     = (float*)d_out;

  // ---- convert inputs to bf16 (A-operands plain; B-operands transposed)
  {
    long n = (long)NNODES * INC;
    cvt_f32_bf16_kernel<<<(n + 255) / 256, 256, 0, stream>>>(x, xbf, n);
    long ng = (long)HEADS * INC * HID;
    cvt_f32_bf16_T_kernel<<<(ng + 255) / 256, 256, 0, stream>>>(W_gat, WgatT, INC, HID, ng);
    long nc = (long)LAYERS * HID * HID;
    cvt_f32_bf16_T_kernel<<<(nc + 255) / 256, 256, 0, stream>>>(W_gcn, WgcnT, HID, HID, nc);
  }

  // ---- h = x @ W_gat per head : M=4096, N=512, K=512, z-batched over heads
  {
    dim3 grid(HID / GEMM_TN, NNODES / GEMM_TM, HEADS);
    gemm_bf16_wmma<<<grid, 256, 0, stream>>>(
        xbf, WgatT, h_f32, nullptr,
        NNODES, HID, INC, /*lda*/ INC, /*ldbT*/ INC, /*ldc*/ HD,
        /*b_zstride*/ (long)INC * HID, /*c_zcoloff*/ HID, /*relu*/ 0);
  }

  // ---- attention logits + segment softmax + scatter into Hcur (=H0)
  lg_kernel<<<NNODES * HEADS, 128, 0, stream>>>(h_f32, a_src, a_dst, lgs, lgd);
  fill_f32_kernel<<<(NNODES * HEADS + 255) / 256, 256, 0, stream>>>(mbuf, -1e30f, NNODES * HEADS);
  (void)hipMemsetAsync(zbuf, 0, (size_t)NNODES * HEADS * 4, stream);
  edge_e_kernel<<<(NEDGES + 255) / 256, 256, 0, stream>>>(src, dst, lgs, lgd, ew, mbuf);
  edge_w_kernel<<<(NEDGES + 255) / 256, 256, 0, stream>>>(dst, ew, mbuf, zbuf);
  (void)hipMemsetAsync(Hcur, 0, (size_t)NNODES * HD * 4, stream);
  edge_scatter_kernel<<<NEDGES, 256, 0, stream>>>(src, dst, ew, zbuf, h_f32, Hcur);

  // ---- dense A_hat = D^-1/2 (A+I) D^-1/2, stored bf16 (A operand: row-major)
  (void)hipMemsetAsync(A_f32, 0, (size_t)NNODES * NNODES * 4, stream);
  scatter_A_kernel<<<(NEDGES + 255) / 256, 256, 0, stream>>>(src, dst, A_f32);
  row_deg_kernel<<<NNODES, 256, 0, stream>>>(A_f32, dinv);
  {
    long n = (long)NNODES * NNODES;
    ahat_kernel<<<(n + 255) / 256, 256, 0, stream>>>(A_f32, dinv, Ahat);
  }

  // ---- 3 residual GCN layers: Hcur <- relu((A_hat @ Hcur) @ W_l) + Hcur
  for (int l = 0; l < LAYERS; ++l) {
    long nh = (long)NNODES * HD;
    // Hcur [4096,2048] -> HbfT [2048,4096] (B operand pre-transposed)
    cvt_f32_bf16_T_kernel<<<(nh + 255) / 256, 256, 0, stream>>>(Hcur, HbfT, NNODES, HD, nh);

    // P = A_hat @ H : M=4096, N=2048, K=4096
    dim3 g1(HD / GEMM_TN, NNODES / GEMM_TM, 1);
    gemm_bf16_wmma<<<g1, 256, 0, stream>>>(
        Ahat, HbfT, P_f32, nullptr,
        NNODES, HD, NNODES, /*lda*/ NNODES, /*ldbT*/ NNODES, /*ldc*/ HD, 0L, 0, 0);

    // P view [N*H,512] is the A operand -> plain bf16 convert
    cvt_f32_bf16_kernel<<<(nh + 255) / 256, 256, 0, stream>>>(P_f32, Pbf, nh);

    // Hcur = relu(P @ W_l) + Hcur : M=16384, N=512, K=512
    dim3 g2(HID / GEMM_TN, (NNODES * HEADS) / GEMM_TM, 1);
    gemm_bf16_wmma<<<g2, 256, 0, stream>>>(
        Pbf, WgcnT + (long)l * HID * HID, Hcur, /*residual*/ Hcur,
        NNODES * HEADS, HID, HID, /*lda*/ HID, /*ldbT*/ HID, /*ldc*/ HID,
        0L, 0, /*relu*/ 1);
  }

  // ---- per-graph mean pool, mean over heads, final FC
  (void)hipMemsetAsync(pooled, 0, (size_t)NGRAPH * HD * 4, stream);
  (void)hipMemsetAsync(cnt, 0, (size_t)NGRAPH * 4, stream);
  cnt_kernel<<<(NNODES + 255) / 256, 256, 0, stream>>>(batch, cnt);
  pool_kernel<<<NNODES, 256, 0, stream>>>(batch, Hcur, pooled);
  final_kernel<<<NGRAPH, 256, 0, stream>>>(pooled, cnt, fc_w, fc_b, out);
}
